// MyEncoder_82016695484605
// MI455X (gfx1250) — matrix operand order repaired and checked by reading, not compiled, not run on hardware
//
#include <hip/hip_runtime.h>
#include <math.h>

// ---------------------------------------------------------------------------
// Problem constants (from reference setup_inputs)
// ---------------------------------------------------------------------------
#define B_DIM   2
#define N_DIM   8
#define H_DIM   128
#define D0      128
#define D1      128
#define D2      128
#define C_DIM   3
#define NNZ     2097152           // 2^21
#define XZ      (D0 * D2)         // 16384
#define ROWS    (D0 * D1)         // 16384
#define VOLSZ   (D0 * D1 * D2)    // 2^21
#define BN      (B_DIM * N_DIM)   // 16
#define OUTSZ   (B_DIM * N_DIM * C_DIM * D0 * D1)  // 786432

// exponnorm parameters
#define EN_K    1.790987517302874f
#define EN_LOC  -0.08208043639065216f
#define EN_SCL  0.05720079131427573f

typedef __attribute__((ext_vector_type(16))) _Float16 v16h;
typedef __attribute__((ext_vector_type(8)))  float    v8f;

// ---------------------------------------------------------------------------
// K1: light = sigmoid(light_pattern); light /= frobenius_norm(light)
//     Write as f16 panel lightN[n*128 + h], n in [0,16) (rows 8..15 zero pad).
//     Single block of 256 threads; 1024 live elements.
// ---------------------------------------------------------------------------
__global__ void k1_light_prep(const float* __restrict__ light_pattern,
                              _Float16* __restrict__ lightN)
{
    __shared__ float red[256];
    const int t = threadIdx.x;

    float sv[4];
    float ss = 0.0f;
#pragma unroll
    for (int k = 0; k < 4; ++k) {
        const int i = t + 256 * k;                 // i < 1024 = N*H
        const float x = light_pattern[i];
        const float s = 1.0f / (1.0f + __expf(-x));
        sv[k] = s;
        ss += s * s;
    }
    red[t] = ss;
    __syncthreads();
#pragma unroll
    for (int off = 128; off > 0; off >>= 1) {
        if (t < off) red[t] += red[t + off];
        __syncthreads();
    }
    const float invn = 1.0f / sqrtf(red[0]);

#pragma unroll
    for (int k = 0; k < 4; ++k) {
        const int i = t + 256 * k;
        lightN[i]        = (_Float16)(sv[k] * invn); // rows n=0..7
        lightN[1024 + i] = (_Float16)0.0f;           // zero pad rows n=8..15
    }
}

// ---------------------------------------------------------------------------
// K2: WMMA GEMM  light_buf[xz, n] = sum_h light_info[h, xz] * lightN[n, h]
//     D(16x16 f32) = A(16x32 f16: M=xz tile, K=h) x B(32x16 f16: K=h, N=n)
//     One wave32 per 16-row xz tile; 4 K-chunks of 32 cover H=128.
//     A/B use contiguous k = 16*hi + e (same in both); D layout per ISA 7.12.2.
// ---------------------------------------------------------------------------
__global__ void k2_light_gemm(const float* __restrict__ light_info,
                              const _Float16* __restrict__ lightN,
                              float* __restrict__ light_buf)
{
    const int wave   = blockIdx.x * (blockDim.x >> 5) + (threadIdx.x >> 5);
    const int lane   = threadIdx.x & 31;
    const int xzBase = wave * 16;                  // 1024 tiles cover XZ=16384
    const int m      = lane & 15;                  // row for A, col N for B/D
    const int hi     = lane >> 4;                  // half-wave selector

    v8f c = {};
#pragma unroll
    for (int hBase = 0; hBase < H_DIM; hBase += 32) {
        // A tile: A[m][k] = light_info[(hBase+k)*XZ + xzBase + m]
        // A k order used here (contiguous, not the ISA table order): lanes 0-15 K=0..15, lanes 16-31 K=16..31 (matches B)
        v16h a;
#pragma unroll
        for (int e = 0; e < 16; ++e) {
            const int k = 16 * hi + e;
            a[e] = (_Float16)light_info[(hBase + k) * XZ + xzBase + m];
        }
        // B tile: lanes 0-15 hold K=0..15 of column N=lane, lanes 16-31 K=16..31
        const v16h b = *(const v16h*)(lightN + m * H_DIM + hBase + 16 * hi);

        c = __builtin_amdgcn_wmma_f32_16x16x32_f16(
                /*neg_a=*/false, a, /*neg_b=*/false, b,
                /*c_mod=*/(short)0, c, /*reuse_a=*/false, /*reuse_b=*/false);
    }

    // D layout: lanes 0-15 VGPR v -> (M=v, N=lane); lanes 16-31 -> (M=v+8)
#pragma unroll
    for (int v = 0; v < 8; ++v) {
        const int mm = v + 8 * hi;
        light_buf[(xzBase + mm) * BN + m] = c[v];
    }
}

// ---------------------------------------------------------------------------
// K3: zero the (C, ROWS, BN) f32 accumulator (atomic target). float4 stores.
// ---------------------------------------------------------------------------
__global__ void k3_zero_acc(float4* __restrict__ acc4)
{
    const int i = blockIdx.x * blockDim.x + threadIdx.x;   // 196608 float4
    acc4[i] = make_float4(0.f, 0.f, 0.f, 0.f);
}

// ---------------------------------------------------------------------------
// K4: fused gather * density * light -> atomic scatter into acc.
//     One thread per (nonzero, output-column j), j = b*8 + n, 16 per nonzero.
//     density_volume (16.8MB) and light_buf (1MB) are L2-resident;
//     per-thread: one coalesced light load, one broadcast density load,
//     one global_atomic_add_f32.
// ---------------------------------------------------------------------------
__global__ void k4_scatter(const int*   __restrict__ ray_rows,
                           const int*   __restrict__ ray_cols,
                           const float* __restrict__ ray_vals,
                           const float* __restrict__ density,
                           const float* __restrict__ light_buf,
                           float*       __restrict__ acc)
{
    const unsigned gid = blockIdx.x * blockDim.x + threadIdx.x; // < C*NNZ*16
    const unsigned i   = gid >> 4;     // flat nonzero index over (C, NNZ)
    const int      j   = gid & 15;     // output column = b*8 + n
    const int      b   = j >> 3;
    // const int   n   = j & 7;        // == light column selector below

    // prefetch upcoming index stream (emits global_prefetch_b8)
    if (i + 4096u < (unsigned)(C_DIM * NNZ)) {
        __builtin_prefetch(&ray_cols[i + 4096u], 0, 1);
        __builtin_prefetch(&ray_rows[i + 4096u], 0, 1);
    }

    const int   row = ray_rows[i];
    const int   col = ray_cols[i];            // flat (X,Y,Z), Z fastest
    const float val = ray_vals[i];

    const int x  = col >> 14;                 // col / (D1*D2)
    const int z  = col & 127;                 // col % D2
    const int xz = (x << 7) | z;              // x*128 + z

    const float den = density[(b << 21) + col];        // density[b, col]
    const float lt  = light_buf[xz * BN + (j & 7)];    // light[n, x, z]

    const int c = (int)(i >> 21);             // camera = i / NNZ
    atomicAdd(&acc[(c * ROWS + row) * BN + j], val * den * lt);
}

// ---------------------------------------------------------------------------
// K5: transpose acc (C,ROWS,BN) -> out (B,N,C,D0,D1) and apply exponnorm
//     noise (deterministic per-element hash RNG; Box-Muller + -log(u)).
// ---------------------------------------------------------------------------
__device__ __forceinline__ unsigned pcg_hash(unsigned x)
{
    x = x * 747796405u + 2891336453u;
    x = ((x >> ((x >> 28u) + 4u)) ^ x) * 277803737u;
    return (x >> 22u) ^ x;
}

__global__ void k5_finalize(const float* __restrict__ acc,
                            float* __restrict__ out)
{
    const int o = blockIdx.x * blockDim.x + threadIdx.x;   // < OUTSZ
    const int r  = o & (ROWS - 1);
    const int t  = o >> 14;          // (b*8+n)*3 + c
    const int c  = t % 3;
    const int bn = t / 3;            // b*8 + n

    const float v = acc[(c * ROWS + r) * BN + bn];

    // deterministic exponnorm-style noise: loc + scale*N(0,1) + K*scale*Exp(1)
    const unsigned h1 = pcg_hash((unsigned)o * 3u + 0x9E3779B9u);
    const unsigned h2 = pcg_hash((unsigned)o * 3u + 0x85EBCA6Bu);
    const unsigned h3 = pcg_hash((unsigned)o * 3u + 0xC2B2AE35u);
    const float u1 = ((h1 >> 8) + 1u) * (1.0f / 16777216.0f);  // (0,1]
    const float u2 = (h2 >> 8)        * (1.0f / 16777216.0f);  // [0,1)
    const float u3 = ((h3 >> 8) + 1u) * (1.0f / 16777216.0f);  // (0,1]
    const float nrm = sqrtf(-2.0f * __logf(u1)) * __cosf(6.28318530718f * u2);
    const float exp1 = -__logf(u3);
    const float noise = EN_LOC + EN_SCL * nrm + EN_K * EN_SCL * exp1;

    out[o] = v + v * noise;
}

// ---------------------------------------------------------------------------
// Host launcher
// ---------------------------------------------------------------------------
extern "C" void kernel_launch(void* const* d_in, const int* in_sizes, int n_in,
                              void* d_out, int out_size, void* d_ws, size_t ws_size,
                              hipStream_t stream)
{
    (void)in_sizes; (void)n_in; (void)out_size; (void)ws_size;

    const float* density       = (const float*)d_in[0]; // (2,128,128,128)
    const float* light_pattern = (const float*)d_in[1]; // (8,128)
    const float* light_info    = (const float*)d_in[2]; // (128,128,128)
    const int*   ray_rows      = (const int*)  d_in[3]; // (3, NNZ)
    const int*   ray_cols      = (const int*)  d_in[4]; // (3, NNZ)
    const float* ray_vals      = (const float*)d_in[5]; // (3, NNZ)
    float*       out           = (float*)d_out;         // (2,8,3,128,128) f32

    // workspace layout
    _Float16* lightN    = (_Float16*)d_ws;                       // 16*128 f16 (4KB)
    float*    light_buf = (float*)((char*)d_ws + 4096);          // 16384*16 f32 (1MB)
    float*    acc       = light_buf + (size_t)XZ * BN;           // 3*16384*16 f32 (3MB)

    // K1: sigmoid + global Frobenius normalization -> f16 B panel
    k1_light_prep<<<1, 256, 0, stream>>>(light_pattern, lightN);

    // K2: WMMA GEMM -> light_buf[xz, n].  1024 tiles, 8 waves/block.
    k2_light_gemm<<<128, 256, 0, stream>>>(light_info, lightN, light_buf);

    // K3: zero accumulator (196608 float4)
    k3_zero_acc<<<768, 256, 0, stream>>>((float4*)acc);

    // K4: fused gather/scale/atomic scatter.  C*NNZ*16 threads.
    k4_scatter<<<(C_DIM * NNZ * 16) / 256, 256, 0, stream>>>(
        ray_rows, ray_cols, ray_vals, density, light_buf, acc);

    // K5: transpose + noise -> d_out
    k5_finalize<<<OUTSZ / 256, 256, 0, stream>>>(acc, out);
}
